// SentenceVAE_41953240547837
// MI455X (gfx1250) — compile-verified, matrix operand order checked by reading, not run on hardware
//
#include <hip/hip_runtime.h>
#include <hip/hip_bf16.h>

// Problem constants (match reference)
#define VV 32000
#define EE 256
#define HH 512
#define LL 16
#define BB 32
#define SS 60
#define G3 1536   // 3*H

typedef __attribute__((ext_vector_type(16))) __bf16 v16bf;
typedef __attribute__((ext_vector_type(8)))  __bf16 v8bf;
typedef __attribute__((ext_vector_type(8)))  float  v8f;

// ---------------------------------------------------------------------------
// Operand loaders for V_WMMA_F32_16X16X32_BF16 (ISA 7.12.2, 16-bit layouts).
// A (16xK) and B (Kx16, stored N-major i.e. transposed) use the same mapping:
//   lane in [0,16): row = lane, half = 0 ; lane in [16,32): row = lane-16, half=1
//   elements 0..7  <- K = kbase + half*8 + [0..7]
//   elements 8..15 <- K = kbase + 16 + half*8 + [0..7]
// ---------------------------------------------------------------------------
__device__ inline v16bf load_bf_tile(const __bf16* base, int ldk, int lane) {
  int m = lane & 15, half = lane >> 4;
  const __bf16* p = base + (size_t)m * ldk + half * 8;
  v8bf lo = *(const v8bf*)(p);
  v8bf hi = *(const v8bf*)(p + 16);
  return __builtin_shufflevector(lo, hi, 0,1,2,3,4,5,6,7,8,9,10,11,12,13,14,15);
}

// Same mapping but source is fp32 (LDS-resident hidden state), convert in-reg.
__device__ inline v16bf load_a_f32(const float* base, int ldk, int lane) {
  int m = lane & 15, half = lane >> 4;
  const float* p = base + (size_t)m * ldk + half * 8;
  v16bf a;
#pragma unroll
  for (int i = 0; i < 8; ++i) a[i] = (__bf16)p[i];
#pragma unroll
  for (int i = 0; i < 8; ++i) a[8 + i] = (__bf16)p[16 + i];
  return a;
}

__device__ inline v8f wmma_bf16(v16bf a, v16bf b, v8f c) {
  return __builtin_amdgcn_wmma_f32_16x16x32_bf16(false, a, false, b, (short)0, c,
                                                 false, false);
}

// ---------------------------------------------------------------------------
// 0) Stable descending argsort of lengths (B=32) == jnp.argsort(-length)
// ---------------------------------------------------------------------------
__global__ void prep_sort(const int* __restrict__ len, int* sorted_idx, int* slen) {
  if (threadIdx.x == 0) {
    int idx[BB];
    for (int i = 0; i < BB; ++i) idx[i] = i;
    for (int i = 1; i < BB; ++i) {          // stable insertion sort, descending
      int key = idx[i], kl = len[key], j = i - 1;
      while (j >= 0 && len[idx[j]] < kl) { idx[j + 1] = idx[j]; --j; }
      idx[j + 1] = key;
    }
    for (int i = 0; i < BB; ++i) { sorted_idx[i] = idx[i]; slen[i] = len[idx[i]]; }
  }
}

// ---------------------------------------------------------------------------
// 1) Tiled transpose + fp32->bf16 convert:  in[R][C] -> out[C][R]
// ---------------------------------------------------------------------------
__global__ __launch_bounds__(256) void transpose_cvt(const float* __restrict__ in,
                                                     __bf16* __restrict__ out,
                                                     int R, int C) {
  __shared__ float tile[32][33];
  int x = blockIdx.x * 32 + threadIdx.x;
#pragma unroll
  for (int j = 0; j < 32; j += 8) {
    int y = blockIdx.y * 32 + threadIdx.y + j;
    if (y < R && x < C) tile[threadIdx.y + j][threadIdx.x] = in[(size_t)y * C + x];
  }
  __syncthreads();
  int xo = blockIdx.y * 32 + threadIdx.x;
#pragma unroll
  for (int j = 0; j < 32; j += 8) {
    int yo = blockIdx.x * 32 + threadIdx.y + j;
    if (yo < C && xo < R)
      out[(size_t)yo * R + xo] = (__bf16)tile[threadIdx.x][threadIdx.y + j];
  }
}

// ---------------------------------------------------------------------------
// 2) GX = gather(embedding[tokens_sorted]) @ Wih + bih   (bf16 WMMA, fp32 out)
//    rows = (sorted_b * S + t), 1920 x 1536, K = 256.
//    mode: pos_rep/neg_rep non-null => decoder sentiment substitution.
// ---------------------------------------------------------------------------
__global__ __launch_bounds__(256) void gx_kernel(
    const int* __restrict__ tokens, const float* __restrict__ emb,
    const __bf16* __restrict__ WihT, const float* __restrict__ bih,
    const int* __restrict__ sorted_idx, const float* __restrict__ pos_rep,
    const float* __restrict__ neg_rep, float* __restrict__ GX) {
  __shared__ __bf16 Abuf[16][EE];  // 16-row A tile, bf16, 8 KB
  int tid = threadIdx.x;
  int mt = blockIdx.x;  // 0..119
  for (int idx = tid; idx < 16 * EE; idx += 256) {
    int r = idx >> 8, e = idx & (EE - 1);
    int row = mt * 16 + r;
    int bs = row / SS, t = row - bs * SS;
    int tok = tokens[sorted_idx[bs] * SS + t];
    float v;
    if (pos_rep && tok == 10)      v = pos_rep[e];
    else if (neg_rep && tok == 11) v = neg_rep[e];
    else                           v = emb[(size_t)tok * EE + e];
    Abuf[r][e] = (__bf16)v;
  }
  __syncthreads();
  int wave = tid >> 5, lane = tid & 31;
  for (int nt = wave; nt < G3 / 16; nt += 8) {
    v8f acc = {};
#pragma unroll
    for (int kc = 0; kc < EE / 32; ++kc) {
      v16bf a = load_bf_tile(&Abuf[0][kc * 32], EE, lane);
      v16bf b = load_bf_tile(WihT + (size_t)(nt * 16) * EE + kc * 32, EE, lane);
      acc = wmma_bf16(a, b, acc);
    }
    int half = lane >> 4, col = nt * 16 + (lane & 15);
    float bv = bih[col];
#pragma unroll
    for (int r = 0; r < 8; ++r) {
      int row = mt * 16 + r + 8 * half;
      GX[(size_t)row * G3 + col] = acc[r] + bv;
    }
  }
}

// ---------------------------------------------------------------------------
// 3) Persistent single-workgroup GRU. h in LDS (fp32, 64 KB of 320 KB/WGP).
//    Per step: gh = h @ WhhT^T via bf16 WMMA (32 waves x 2 col-groups x 3 gates),
//    then fp32 VALU gate math, masked update (PyTorch packed-sequence semantics).
// ---------------------------------------------------------------------------
__global__ __launch_bounds__(1024) void gru_kernel(
    const float* __restrict__ GX, const __bf16* __restrict__ WhhT,
    const float* __restrict__ bhh, const int* __restrict__ slen,
    const int* __restrict__ sorted_idx, const float* __restrict__ h0,
    float* __restrict__ h_last, __bf16* __restrict__ y_out) {
  __shared__ float hs[BB * HH];  // 64 KB
  int tid = threadIdx.x, wave = tid >> 5, lane = tid & 31;
  int half = lane >> 4, n = lane & 15;
  for (int i = tid; i < BB * HH; i += 1024) hs[i] = h0 ? h0[i] : 0.0f;
  __syncthreads();

#pragma unroll 1
  for (int t = 0; t < SS; ++t) {
    v8f acc[2][3];
    // ---- compute phase: gh tiles for this wave's 2 groups ----
#pragma unroll 1
    for (int g = 0; g < 2; ++g) {
      int grp = wave * 2 + g;
      int mt = grp >> 5, jt = grp & 31;
      const __bf16* wr = WhhT + (size_t)(0 * HH + jt * 16) * HH;
      const __bf16* wz = WhhT + (size_t)(1 * HH + jt * 16) * HH;
      const __bf16* wn = WhhT + (size_t)(2 * HH + jt * 16) * HH;
      v8f ar = {}, az = {}, an = {};
#pragma unroll 1
      for (int kc = 0; kc < HH / 32; ++kc) {
        // prefetch next K-chunk of the streamed weights (latency-critical loop)
        __builtin_prefetch(wr + (kc + 1) * 32, 0, 1);
        __builtin_prefetch(wz + (kc + 1) * 32, 0, 1);
        __builtin_prefetch(wn + (kc + 1) * 32, 0, 1);
        v16bf a  = load_a_f32(&hs[(mt * 16) * HH + kc * 32], HH, lane);
        v16bf br = load_bf_tile(wr + kc * 32, HH, lane);
        v16bf bz = load_bf_tile(wz + kc * 32, HH, lane);
        v16bf bn = load_bf_tile(wn + kc * 32, HH, lane);
        ar = wmma_bf16(a, br, ar);
        az = wmma_bf16(a, bz, az);
        an = wmma_bf16(a, bn, an);
      }
      acc[g][0] = ar; acc[g][1] = az; acc[g][2] = an;
    }
    __syncthreads();  // all A-reads of hs done before anyone writes hs
    // ---- update phase: gates + masked state update ----
#pragma unroll 1
    for (int g = 0; g < 2; ++g) {
      int grp = wave * 2 + g;
      int mt = grp >> 5, jt = grp & 31;
      int col = jt * 16 + n;
      float br_ = bhh[col], bz_ = bhh[HH + col], bn_ = bhh[2 * HH + col];
#pragma unroll
      for (int r = 0; r < 8; ++r) {
        int b = mt * 16 + r + 8 * half;
        const float* gx = GX + (size_t)(b * SS + t) * G3;
        float ghr = acc[g][0][r] + br_;
        float ghz = acc[g][1][r] + bz_;
        float ghn = acc[g][2][r] + bn_;
        float rg = 1.0f / (1.0f + __expf(-(gx[col] + ghr)));
        float zg = 1.0f / (1.0f + __expf(-(gx[HH + col] + ghz)));
        float ng = tanhf(gx[2 * HH + col] + rg * ghn);
        float hold = hs[b * HH + col];
        float hnew = (1.0f - zg) * ng + zg * hold;
        bool mv = t < slen[b];
        hs[b * HH + col] = mv ? hnew : hold;
        if (y_out)  // decoder: write y (unsorted row order, padded -> 0) as bf16
          y_out[((size_t)sorted_idx[b] * SS + t) * HH + col] =
              (__bf16)(mv ? hnew : 0.0f);
      }
    }
    __syncthreads();  // hs updates visible before next step's A loads
  }
  if (h_last)
    for (int i = tid; i < BB * HH; i += 1024) h_last[i] = hs[i];
}

// ---------------------------------------------------------------------------
// 4) Latent head: mean/logv/z (sorted order, as the reference returns them)
//    and decoder initial hidden = z @ W_l2h + b_l2h. Tiny -> fp32 VALU.
// ---------------------------------------------------------------------------
__global__ __launch_bounds__(512) void latent_kernel(
    const float* __restrict__ h_last, const float* __restrict__ W_mean,
    const float* __restrict__ b_mean, const float* __restrict__ W_logv,
    const float* __restrict__ b_logv, const float* __restrict__ eps,
    const float* __restrict__ W_l2h, const float* __restrict__ b_l2h,
    float* __restrict__ hidden, float* __restrict__ out_tail) {
  __shared__ float zsh[BB * LL];
  int tid = threadIdx.x;            // 512 == B*L
  int b = tid >> 4, l = tid & 15;
  float sm = b_mean[l], sv = b_logv[l];
  for (int k = 0; k < HH; ++k) {
    float h = h_last[b * HH + k];
    sm += h * W_mean[k * LL + l];
    sv += h * W_logv[k * LL + l];
  }
  float zz = eps[b * LL + l] * __expf(0.5f * sv) + sm;
  out_tail[tid] = sm;               // mean  [B,L]
  out_tail[512 + tid] = sv;         // logv  [B,L]
  out_tail[1024 + tid] = zz;        // z     [B,L]
  zsh[tid] = zz;
  __syncthreads();
  for (int i = tid; i < BB * HH; i += 512) {
    int bb = i >> 9, c = i & (HH - 1);
    float s = b_l2h[c];
#pragma unroll
    for (int l2 = 0; l2 < LL; ++l2) s += zsh[bb * LL + l2] * W_l2h[l2 * HH + c];
    hidden[i] = s;
  }
}

// ---------------------------------------------------------------------------
// 5) Output projection: logits[1920,32000] = A[1920,512](bf16) @ W_out + b_out.
//    2Mx2N register blocking per wave: 4 WMMA per 4 operand loads (2x the
//    flop/byte of 1x1), so the loop tracks the WMMA ceiling, not load issue.
//    Block = 8 waves -> 32 rows x 256 cols per block.
// ---------------------------------------------------------------------------
__global__ __launch_bounds__(256) void proj_kernel(
    const __bf16* __restrict__ A, const __bf16* __restrict__ WoutT,
    const float* __restrict__ bias, float* __restrict__ out) {
  int mt2 = blockIdx.x;                      // 0..59   (32 rows each)
  int wave = threadIdx.x >> 5, lane = threadIdx.x & 31;
  int nt0 = (blockIdx.y * 8 + wave) * 2;     // 0..1998 (two 16-col tiles)
  const __bf16* abase0 = A + (size_t)(mt2 * 32) * HH;
  const __bf16* abase1 = abase0 + (size_t)16 * HH;
  const __bf16* bbase0 = WoutT + (size_t)(nt0 * 16) * HH;
  const __bf16* bbase1 = bbase0 + (size_t)16 * HH;
  v8f acc00 = {}, acc01 = {}, acc10 = {}, acc11 = {};
#pragma unroll
  for (int kc = 0; kc < HH / 32; ++kc) {
    __builtin_prefetch(bbase0 + (kc + 4) * 32, 0, 1);  // stream-ahead on W_out
    __builtin_prefetch(bbase1 + (kc + 4) * 32, 0, 1);
    v16bf a0 = load_bf_tile(abase0 + kc * 32, HH, lane);
    v16bf a1 = load_bf_tile(abase1 + kc * 32, HH, lane);
    v16bf b0 = load_bf_tile(bbase0 + kc * 32, HH, lane);
    v16bf b1 = load_bf_tile(bbase1 + kc * 32, HH, lane);
    acc00 = wmma_bf16(a0, b0, acc00);
    acc01 = wmma_bf16(a0, b1, acc01);
    acc10 = wmma_bf16(a1, b0, acc10);
    acc11 = wmma_bf16(a1, b1, acc11);
  }
  int half = lane >> 4, n = lane & 15;
  int col0 = nt0 * 16 + n, col1 = col0 + 16;
  float bo0 = bias[col0], bo1 = bias[col1];
#pragma unroll
  for (int r = 0; r < 8; ++r) {
    int row0 = mt2 * 32 + r + 8 * half;
    int row1 = row0 + 16;
    out[(size_t)row0 * VV + col0] = acc00[r] + bo0;
    out[(size_t)row0 * VV + col1] = acc01[r] + bo1;
    out[(size_t)row1 * VV + col0] = acc10[r] + bo0;
    out[(size_t)row1 * VV + col1] = acc11[r] + bo1;
  }
}

// ---------------------------------------------------------------------------
// 6) Per-row online logsumexp over V=32000, then 7) subtract (log_softmax).
// ---------------------------------------------------------------------------
__global__ __launch_bounds__(256) void lse_kernel(const float* __restrict__ logits,
                                                  float* __restrict__ lse) {
  __shared__ float smx[256], ssm[256];
  const float* p = logits + (size_t)blockIdx.x * VV;
  float m = -3.4e38f, s = 0.0f;
  for (int i = threadIdx.x; i < VV; i += 256) {
    float x = p[i];
    float nm = fmaxf(m, x);
    s = s * __expf(m - nm) + __expf(x - nm);
    m = nm;
  }
  smx[threadIdx.x] = m; ssm[threadIdx.x] = s;
  __syncthreads();
  for (int st = 128; st > 0; st >>= 1) {
    if (threadIdx.x < st) {
      float m2 = smx[threadIdx.x + st], s2 = ssm[threadIdx.x + st];
      float nm = fmaxf(smx[threadIdx.x], m2);
      ssm[threadIdx.x] = ssm[threadIdx.x] * __expf(smx[threadIdx.x] - nm) +
                         s2 * __expf(m2 - nm);
      smx[threadIdx.x] = nm;
    }
    __syncthreads();
  }
  if (threadIdx.x == 0) lse[blockIdx.x] = smx[0] + __logf(ssm[0]);
}

// row = blockIdx.y; each block strips 1024 cols as float4 (no divides).
__global__ __launch_bounds__(256) void sub_kernel(float* __restrict__ out,
                                                  const float* __restrict__ lse) {
  int row = blockIdx.y;
  int col = blockIdx.x * 1024 + threadIdx.x * 4;
  if (col < VV) {
    float l = lse[row];
    float4* p = (float4*)(out + (size_t)row * VV + col);
    float4 v = *p;
    v.x -= l; v.y -= l; v.z -= l; v.w -= l;
    *p = v;
  }
}

// ---------------------------------------------------------------------------
// Host launcher
// ---------------------------------------------------------------------------
extern "C" void kernel_launch(void* const* d_in, const int* in_sizes, int n_in,
                              void* d_out, int out_size, void* d_ws, size_t ws_size,
                              hipStream_t stream) {
  const int*   seq      = (const int*)d_in[0];
  const int*   len      = (const int*)d_in[1];
  const int*   dec      = (const int*)d_in[2];
  const float* emb      = (const float*)d_in[3];
  const float* enc_Wih  = (const float*)d_in[4];
  const float* enc_Whh  = (const float*)d_in[5];
  const float* enc_bih  = (const float*)d_in[6];
  const float* enc_bhh  = (const float*)d_in[7];
  const float* dec_Wih  = (const float*)d_in[8];
  const float* dec_Whh  = (const float*)d_in[9];
  const float* dec_bih  = (const float*)d_in[10];
  const float* dec_bhh  = (const float*)d_in[11];
  const float* W_mean   = (const float*)d_in[12];
  const float* b_mean   = (const float*)d_in[13];
  const float* W_logv   = (const float*)d_in[14];
  const float* b_logv   = (const float*)d_in[15];
  const float* W_l2h    = (const float*)d_in[16];
  const float* b_l2h    = (const float*)d_in[17];
  const float* W_out    = (const float*)d_in[18];
  const float* b_out    = (const float*)d_in[19];
  const float* pos_rep  = (const float*)d_in[20];
  const float* neg_rep  = (const float*)d_in[21];
  const float* eps      = (const float*)d_in[22];

  char* ws = (char*)d_ws;
  size_t off = 0;
  auto alloc = [&](size_t bytes) -> void* {
    void* p = ws + off;
    off = (off + bytes + 255) & ~(size_t)255;
    return p;
  };
  int*    sorted_idx = (int*)alloc(BB * sizeof(int));
  int*    slen       = (int*)alloc(BB * sizeof(int));
  __bf16* WihT_e     = (__bf16*)alloc((size_t)G3 * EE * 2);
  __bf16* WihT_d     = (__bf16*)alloc((size_t)G3 * EE * 2);
  __bf16* WhhT_e     = (__bf16*)alloc((size_t)G3 * HH * 2);
  __bf16* WhhT_d     = (__bf16*)alloc((size_t)G3 * HH * 2);
  __bf16* WoutT      = (__bf16*)alloc((size_t)VV * HH * 2);
  float*  GXe        = (float*)alloc((size_t)BB * SS * G3 * 4);
  float*  GXd        = (float*)alloc((size_t)BB * SS * G3 * 4);
  float*  h_last     = (float*)alloc((size_t)BB * HH * 4);
  float*  hidden     = (float*)alloc((size_t)BB * HH * 4);
  __bf16* A_proj     = (__bf16*)alloc((size_t)BB * SS * HH * 2);
  float*  lse        = (float*)alloc((size_t)BB * SS * 4);

  float* logits   = (float*)d_out;                                  // [B*S, V]
  float* out_tail = (float*)d_out + (size_t)BB * SS * VV;           // mean|logv|z

  prep_sort<<<1, 32, 0, stream>>>(len, sorted_idx, slen);

  dim3 tb(32, 8);
  transpose_cvt<<<dim3(G3 / 32, EE / 32), tb, 0, stream>>>(enc_Wih, WihT_e, EE, G3);
  transpose_cvt<<<dim3(G3 / 32, EE / 32), tb, 0, stream>>>(dec_Wih, WihT_d, EE, G3);
  transpose_cvt<<<dim3(G3 / 32, HH / 32), tb, 0, stream>>>(enc_Whh, WhhT_e, HH, G3);
  transpose_cvt<<<dim3(G3 / 32, HH / 32), tb, 0, stream>>>(dec_Whh, WhhT_d, HH, G3);
  transpose_cvt<<<dim3(VV / 32, HH / 32), tb, 0, stream>>>(W_out, WoutT, HH, VV);

  // Encoder / decoder input-side GEMMs (hoisted out of the recurrences)
  gx_kernel<<<(BB * SS) / 16, 256, 0, stream>>>(seq, emb, WihT_e, enc_bih,
                                                sorted_idx, nullptr, nullptr, GXe);
  gx_kernel<<<(BB * SS) / 16, 256, 0, stream>>>(dec, emb, WihT_d, dec_bih,
                                                sorted_idx, pos_rep, neg_rep, GXd);

  // Encoder recurrence -> h_last
  gru_kernel<<<1, 1024, 0, stream>>>(GXe, WhhT_e, enc_bhh, slen, sorted_idx,
                                     nullptr, h_last, nullptr);
  // Latent head (writes mean/logv/z to d_out tail, hidden for decoder)
  latent_kernel<<<1, 512, 0, stream>>>(h_last, W_mean, b_mean, W_logv, b_logv,
                                       eps, W_l2h, b_l2h, hidden, out_tail);
  // Decoder recurrence -> A_proj (bf16, original batch order, padded rows = 0)
  gru_kernel<<<1, 1024, 0, stream>>>(GXd, WhhT_d, dec_bhh, slen, sorted_idx,
                                     hidden, nullptr, A_proj);

  // Output projection + log_softmax
  proj_kernel<<<dim3((BB * SS) / 32, VV / (16 * 8 * 2)), 256, 0, stream>>>(
      A_proj, WoutT, b_out, logits);
  lse_kernel<<<BB * SS, 256, 0, stream>>>(logits, lse);
  sub_kernel<<<dim3((VV + 1023) / 1024, BB * SS), 256, 0, stream>>>(logits, lse);
}
